// IteratedLinearNet_23639499997809
// MI455X (gfx1250) — matrix operand order repaired and checked by reading, not compiled, not run on hardware
//
#include <hip/hip_runtime.h>
#include <stdint.h>

// ---------------------------------------------------------------------------
// IteratedLinearNet on MI455X (gfx1250):  h <- h @ W^T, 60 times.
// Compute-bound (AI ~25k FLOP/byte): v_wmma_f32_16x16x32_bf16 main loop.
// 64x64 wave tile (16 WMMA : 16 ds_load_b128), LDS double buffering fed by
// GLOBAL_LOAD_ASYNC_TO_LDS_B128 (ASYNCcnt) overlapping next-tile loads with
// current-tile WMMAs. Round 3: peeled final K-iteration (straight-line loop
// body) and 32-bit epilogue addressing.
// ---------------------------------------------------------------------------

typedef __attribute__((ext_vector_type(16))) __bf16 v16bf;
typedef __attribute__((ext_vector_type(8)))  __bf16 v8bf;
typedef __attribute__((ext_vector_type(4)))  __bf16 v4bf;
typedef __attribute__((ext_vector_type(8)))  float  v8f;

#define GRIDN 2048
#define BATCHM 4096
#define STEPS 60
#define BM 256          // block rows   (4 waves x 64)
#define BN 128          // block cols   (2 waves x 64)
#define BK 32           // one WMMA K-step per LDS tile

union FragU { v16bf v; v8bf h[2]; };

// f32 -> bf16 bulk convert (4 elems/thread, 16B loads, 8B stores)
__global__ __launch_bounds__(256) void f32_to_bf16_kernel(
    const float* __restrict__ s, __bf16* __restrict__ d, long n4)
{
  long i = (long)blockIdx.x * blockDim.x + threadIdx.x;
  if (i >= n4) return;
  const float4 f = ((const float4*)s)[i];
  v4bf o;
  o[0] = (__bf16)f.x; o[1] = (__bf16)f.y; o[2] = (__bf16)f.z; o[3] = (__bf16)f.w;
  *(v4bf*)&d[i * 4] = o;
}

// C[M,N] = A[M,K] * B[N,K]^T  (y = h @ W^T), bf16 in, f32 accum.
// 256 threads = 8 waves (4x2); each wave owns a 64x64 tile = 4x4 WMMA accs.
__global__ __launch_bounds__(256) void bf16_gemm_kernel(
    const __bf16* __restrict__ A,   // [M, K]  (h, bf16)
    const __bf16* __restrict__ B,   // [N, K]  (W, bf16)
    __bf16* __restrict__ outB,      // [M, N]  bf16 (intermediate steps)
    float*  __restrict__ outF,      // [M, N]  f32  (final step)
    int write_f32)
{
  // Double-buffered tiles: 2*(256*32 + 128*32)*2B = 48 KB LDS.
  __shared__ __bf16 As[2][BM * BK];
  __shared__ __bf16 Bs[2][BN * BK];

  const int K = GRIDN, N = GRIDN;
  const int tid  = threadIdx.x;
  const int lane = tid & 31;
  const int wave = tid >> 5;       // 0..7 (wave32)
  const int wm   = wave >> 1;      // 0..3 -> 64 rows
  const int wn   = wave & 1;       // 0..1 -> 64 cols
  const int half = lane >> 4;      // 0/1
  const int l16  = lane & 15;
  const size_t m0 = (size_t)blockIdx.y * BM;
  const size_t n0 = (size_t)blockIdx.x * BN;

  v8f acc[4][4] = {};

  // Issue one tile's async global->LDS loads: (BM+BN)*BK bf16 = 24 KB
  // = 1536 x 16B chunks = 6 chunks/thread = 6 ASYNCcnt per wave.
  auto issue_tile = [&](int kt, int buf) {
    #pragma unroll
    for (int i = 0; i < 6; ++i) {
      const int c = tid + i * 256;          // 0..1535 ; c<1024 <=> i<4 (uniform)
      const __bf16* g;
      unsigned l;
      if (c < 1024) {                        // A: 256 rows x 4 chunks
        const int row = c >> 2, ch = c & 3;
        g = &A[(m0 + row) * (size_t)K + kt + ch * 8];
        l = (unsigned)(uintptr_t)&As[buf][row * BK + ch * 8];
      } else {                               // B: 128 rows x 4 chunks
        const int c2 = c - 1024;
        const int row = c2 >> 2, ch = c2 & 3;
        g = &B[(n0 + row) * (size_t)K + kt + ch * 8];
        l = (unsigned)(uintptr_t)&Bs[buf][row * BK + ch * 8];
      }
      asm volatile("global_load_async_to_lds_b128 %0, %1, off"
                   :: "v"(l), "v"(g) : "memory");
    }
  };

  // 16 WMMAs on one LDS tile buffer (B frags resident, A frags cycled).
  auto compute_tile = [&](int buf) {
    FragU b[4];
    #pragma unroll
    for (int ni = 0; ni < 4; ++ni) {
      const int col = wn * 64 + ni * 16 + l16;
      b[ni].h[0] = *(const v8bf*)&Bs[buf][col * BK + half * 16];
      b[ni].h[1] = *(const v8bf*)&Bs[buf][col * BK + half * 16 + 8];
    }
    #pragma unroll
    for (int mi = 0; mi < 4; ++mi) {
      FragU a;
      const int row = wm * 64 + mi * 16 + l16;
      a.h[0] = *(const v8bf*)&As[buf][row * BK + half * 16];
      a.h[1] = *(const v8bf*)&As[buf][row * BK + half * 16 + 8];
      #pragma unroll
      for (int ni = 0; ni < 4; ++ni)
        acc[mi][ni] = __builtin_amdgcn_wmma_f32_16x16x32_bf16(
            /*neg_a=*/false, a.v, /*neg_b=*/false, b[ni].v,
            /*c_mod=*/(short)0, acc[mi][ni],
            /*reuse_a=*/false, /*reuse_b=*/false);
    }
  };

  issue_tile(0, 0);

  // Steady state: straight-line body, last iteration peeled.
  for (int kt = 0; kt < K - BK; kt += BK) {
    const int buf = (kt >> 5) & 1;
    issue_tile(kt + BK, buf ^ 1);
    // Async loads complete in order: <=6 outstanding => previous batch done.
    asm volatile("s_wait_asynccnt 0x6" ::: "memory");
    __syncthreads();   // all waves' current-tile loads landed in LDS
    compute_tile(buf);
    __syncthreads();   // reads done before this buffer is re-targeted
  }
  // Final tile (buffer index (K/BK - 1) & 1 == 1).
  asm volatile("s_wait_asynccnt 0x0" ::: "memory");
  __syncthreads();
  compute_tile(1);

  // Epilogue. D layout: elem v at lane l -> (M = half*8 + v, N = l16).
  // 32-bit offsets (max index 4096*2048 < 2^23).
  const unsigned ebase_row = (unsigned)m0 + wm * 64 + half * 8;
  const unsigned ebase_col = (unsigned)n0 + wn * 64 + l16;
  if (write_f32) {
    #pragma unroll
    for (int mi = 0; mi < 4; ++mi)
      #pragma unroll
      for (int ni = 0; ni < 4; ++ni) {
        const unsigned off0 = (ebase_row + mi * 16) * (unsigned)N
                            + ebase_col + ni * 16;
        #pragma unroll
        for (int v = 0; v < 8; ++v)
          outF[off0 + (unsigned)v * (unsigned)N] = acc[mi][ni][v];
      }
  } else {
    #pragma unroll
    for (int mi = 0; mi < 4; ++mi)
      #pragma unroll
      for (int ni = 0; ni < 4; ++ni) {
        const unsigned off0 = (ebase_row + mi * 16) * (unsigned)N
                            + ebase_col + ni * 16;
        #pragma unroll
        for (int v = 0; v < 8; ++v)
          outB[off0 + (unsigned)v * (unsigned)N] = (__bf16)acc[mi][ni][v];
      }
  }
}

extern "C" void kernel_launch(void* const* d_in, const int* in_sizes, int n_in,
                              void* d_out, int out_size, void* d_ws, size_t ws_size,
                              hipStream_t stream) {
  const float* x = (const float*)d_in[0];   // [4096, 2048]
  const float* W = (const float*)d_in[1];   // [2048, 2048]
  float* out = (float*)d_out;               // [4096, 2048]

  // Workspace (bf16): W (8 MB) | h0 (16 MB) | h1 (16 MB) = 40 MB.
  __bf16* Wb = (__bf16*)d_ws;
  __bf16* h0 = Wb + (size_t)GRIDN * GRIDN;
  __bf16* h1 = h0 + (size_t)BATCHM * GRIDN;

  {
    long n4 = (long)GRIDN * GRIDN / 4;
    f32_to_bf16_kernel<<<(int)((n4 + 255) / 256), 256, 0, stream>>>(W, Wb, n4);
  }
  {
    long n4 = (long)BATCHM * GRIDN / 4;
    f32_to_bf16_kernel<<<(int)((n4 + 255) / 256), 256, 0, stream>>>(x, h0, n4);
  }

  dim3 grid(GRIDN / BN, BATCHM / BM);   // (16, 16) blocks of 256 threads
  __bf16* src = h0;
  __bf16* dst = h1;
  for (int t = 0; t < STEPS; ++t) {
    int last = (t == STEPS - 1);
    bf16_gemm_kernel<<<grid, 256, 0, stream>>>(src, Wb, dst, out, last);
    __bf16* tmp = src; src = dst; dst = tmp;
  }
}
